// EncoderLayer_74071005987259
// MI455X (gfx1250) — matrix operand order repaired and checked by reading, not compiled, not run on hardware
//
#include <hip/hip_runtime.h>
#include <hip/hip_bf16.h>
#include <math.h>

// ---------------------------------------------------------------------------
// Types / WMMA helpers (CDNA5 / gfx1250, wave32)
// ---------------------------------------------------------------------------
typedef _Float16 h8   __attribute__((ext_vector_type(8)));
typedef _Float16 h4   __attribute__((ext_vector_type(4)));
typedef _Float16 v16h __attribute__((ext_vector_type(16)));
typedef float    v8f  __attribute__((ext_vector_type(8)));

#define DEVINL __device__ __forceinline__

DEVINL v16h make16(h8 lo, h8 hi) {
  return __builtin_shufflevector(lo, hi, 0,1,2,3,4,5,6,7,8,9,10,11,12,13,14,15);
}

// A-frag (16x32 f16, MxK): lane&15 = row; halves = K {hi*8+0..7, hi*8+16..23}
DEVINL v16h load_a_frag(const _Float16* base, int stride) {
  const int lane = threadIdx.x & 31;
  const _Float16* p = base + (lane & 15) * stride + ((lane >> 4) << 3);
  h8 lo = *(const h8*)p;
  h8 hi = *(const h8*)(p + 16);
  return make16(lo, hi);
}

// B-frag (32x16 f16, KxN): lane&15 = N col; halves = K {hi*8+0..7, hi*8+16..23}
DEVINL v16h load_b_frag(const _Float16* base, int stride) {
  const int lane = threadIdx.x & 31;
  const _Float16* p = base + (lane & 15) + ((lane >> 4) << 3) * stride;
  v16h r;
#pragma unroll
  for (int s = 0; s < 8; ++s) {
    r[s]     = p[s * stride];
    r[s + 8] = p[(16 + s) * stride];
  }
  return r;
}

DEVINL v8f wmma16(v16h a, v16h b, v8f c) {
  return __builtin_amdgcn_wmma_f32_16x16x32_f16(false, a, false, b,
                                                (short)0, c, false, false);
}

// ---------------------------------------------------------------------------
// CDNA5 async global->LDS copy (ASYNCcnt-tracked, no VGPR round trip).
// LDS operand = low 32 bits of the generic pointer (== AS3 offset; the LDS
// aperture occupies the high 32 bits on gfx1250).
// ---------------------------------------------------------------------------
DEVINL void async_copy_b128(void* lds_dst, const void* gsrc) {
  unsigned loff = (unsigned)(size_t)lds_dst;
  unsigned long long ga = (unsigned long long)(size_t)gsrc;
  asm volatile("global_load_async_to_lds_b128 %0, %1, off"
               :: "v"(loff), "v"(ga) : "memory");
}
DEVINL void wait_async0() {
  asm volatile("s_wait_asynccnt 0" ::: "memory");
}

// ---------------------------------------------------------------------------
// Generic WMMA GEMM:  C(MxN) = A(MxK) * B(KxN) + bias  (+ epilogue)
//   Block tile 128x128, K-step 32, 256 threads (8 waves as 4(M) x 2(N)),
//   each wave: 2 M-frags x 4 N-frags. Double-buffered LDS, ONE barrier per
//   K-step. f16 A staged via async-to-LDS; f32 A / f32 B staged via register
//   prefetch + convert.
// ---------------------------------------------------------------------------
#define EPI_F16_SCALE 0   // out16 = f16((acc+bias)*scale)
#define EPI_F32_RES   1   // out32 = acc+bias+residual
#define EPI_GELU_F16  2   // out16 = f16(gelu_exact(acc+bias))

template<bool AF16, int EPI>
__global__ __launch_bounds__(256) void gemm_wmma(
    const void* __restrict__ Aptr, const float* __restrict__ Bw,
    const float* __restrict__ bias, const float* __restrict__ residual,
    float scale, _Float16* __restrict__ out16, float* __restrict__ out32,
    int M, int N, int K)
{
  __shared__ _Float16 As[2][128 * 40];   // pad stride 40 halves (80B rows, 16B aligned)
  __shared__ _Float16 Bs[2][32 * 136];   // pad stride 136 halves (272B rows, 16B aligned)

  const int tid  = threadIdx.x;
  const int wave = tid >> 5;
  const int lane = tid & 31;
  const int wm   = wave >> 1;            // 0..3  (M)
  const int wn   = wave & 1;             // 0..1  (N)
  const int nlo  = lane & 15;
  const int hi   = lane >> 4;
  const int bM   = blockIdx.y * 128;
  const int bN   = blockIdx.x * 128;
  const int KT   = K >> 5;

  const float*    Af32 = (const float*)Aptr;
  const _Float16* Af16 = (const _Float16*)Aptr;

  v8f acc[2][4];
  const v8f zero8 = {0.f,0.f,0.f,0.f,0.f,0.f,0.f,0.f};
#pragma unroll
  for (int i = 0; i < 2; ++i)
#pragma unroll
    for (int j = 0; j < 4; ++j) acc[i][j] = zero8;

  float4 ra[4];  float4 rb[4];

  auto stageA_async = [&](int kt, int buf) {      // AF16 path: byte-exact copy
#pragma unroll
    for (int j = 0; j < 2; ++j) {
      int lin = tid + 256 * j, row = lin >> 2, c = lin & 3;
      async_copy_b128(&As[buf][row * 40 + c * 8],
                      Af16 + (size_t)(bM + row) * K + kt * 32 + c * 8);
    }
  };
  auto loadA_regs = [&](int kt) {                  // f32 path: needs conversion
#pragma unroll
    for (int j = 0; j < 4; ++j) {
      int lin = tid + 256 * j, row = lin >> 3, c = lin & 7;
      ra[j] = *(const float4*)(Af32 + (size_t)(bM + row) * K + kt * 32 + c * 4);
    }
  };
  auto storeA_regs = [&](int buf) {
#pragma unroll
    for (int j = 0; j < 4; ++j) {
      int lin = tid + 256 * j, row = lin >> 3, c = lin & 7;
      h4 hv = { (_Float16)ra[j].x, (_Float16)ra[j].y,
                (_Float16)ra[j].z, (_Float16)ra[j].w };
      *(h4*)(&As[buf][row * 40 + c * 4]) = hv;
    }
  };
  auto loadB_regs = [&](int kt) {
#pragma unroll
    for (int j = 0; j < 4; ++j) {
      int lin = tid + 256 * j, row = lin >> 5, c = lin & 31;
      rb[j] = *(const float4*)(Bw + (size_t)(kt * 32 + row) * N + bN + c * 4);
    }
  };
  auto storeB_regs = [&](int buf) {
#pragma unroll
    for (int j = 0; j < 4; ++j) {
      int lin = tid + 256 * j, row = lin >> 5, c = lin & 31;
      h4 hv = { (_Float16)rb[j].x, (_Float16)rb[j].y,
                (_Float16)rb[j].z, (_Float16)rb[j].w };
      *(h4*)(&Bs[buf][row * 136 + c * 4]) = hv;
    }
  };

  // ---- prologue: stage tile 0 into buffer 0 ----
  if (AF16) stageA_async(0, 0);
  else      { loadA_regs(0); storeA_regs(0); }
  loadB_regs(0); storeB_regs(0);
  if (AF16) wait_async0();
  __syncthreads();

  for (int kt = 0; kt < KT; ++kt) {
    const int cur = kt & 1, nxt = cur ^ 1;
    const bool hasNext = (kt + 1) < KT;

    // issue next-tile staging early: async A writes LDS[nxt] directly
    // (safe: nxt's readers finished before the barrier that ended iter kt-1)
    if (hasNext) {
      if (AF16) stageA_async(kt + 1, nxt);
      else      loadA_regs(kt + 1);
      loadB_regs(kt + 1);
    }

    // compute on current buffer
    const _Float16* Abase = &As[cur][0] + (wm * 32) * 40;
    v16h a0 = load_a_frag(Abase,           40);
    v16h a1 = load_a_frag(Abase + 16 * 40, 40);
#pragma unroll
    for (int nf = 0; nf < 4; ++nf) {
      v16h b = load_b_frag(&Bs[cur][0] + wn * 64 + nf * 16, 136);
      acc[0][nf] = wmma16(a0, b, acc[0][nf]);
      acc[1][nf] = wmma16(a1, b, acc[1][nf]);
    }

    if (hasNext) {
      if (!AF16) storeA_regs(nxt);
      storeB_regs(nxt);
      if (AF16) wait_async0();
      __syncthreads();           // single barrier per K-step
    }
  }

  // ---- epilogue: C-frag layout -> row = mf*16 + v + 8*hi, col = nf*16 + nlo
#pragma unroll
  for (int mf = 0; mf < 2; ++mf) {
#pragma unroll
    for (int nf = 0; nf < 4; ++nf) {
      const int c = bN + wn * 64 + nf * 16 + nlo;
      const float bb = bias[c];
#pragma unroll
      for (int v = 0; v < 8; ++v) {
        const int r = bM + wm * 32 + mf * 16 + v + 8 * hi;
        float val = acc[mf][nf][v] + bb;
        const size_t idx = (size_t)r * N + c;
        if (EPI == EPI_F16_SCALE) {
          out16[idx] = (_Float16)(val * scale);
        } else if (EPI == EPI_F32_RES) {
          out32[idx] = val + residual[idx];
        } else { // EPI_GELU_F16: exact erf gelu
          float g = 0.5f * val * (1.0f + erff(val * 0.70710678118654752f));
          out16[idx] = (_Float16)g;
        }
      }
    }
  }
}

// ---------------------------------------------------------------------------
// Flash attention: one block per (b, h, 128-query tile); 8 waves x 16 q-rows.
// Streams 64-key blocks with online softmax; Q/V/mask staged via async-to-LDS.
// ---------------------------------------------------------------------------
__global__ __launch_bounds__(256) void attn_kernel(
    const _Float16* __restrict__ q16, const _Float16* __restrict__ k16,
    const _Float16* __restrict__ v16, const unsigned char* __restrict__ amask,
    const unsigned char* __restrict__ pmask, _Float16* __restrict__ o16)
{
  constexpr int S = 2048, Bq = 4, E = 768;
  __shared__ _Float16 Qs[128 * 72];      // [qrow][d], pad 72
  __shared__ _Float16 Ks[64 * 72];       // transposed: [d][key]
  __shared__ _Float16 Vs[64 * 72];       // [key][d]
  __shared__ _Float16 Ps[8 * 16 * 72];   // per-wave P staging [row][key]
  __shared__ uint4    Ms4[128 * 4];      // attn_mask bytes [qrow][64]
  __shared__ uint4    Pk4[4];            // padding mask bytes [64]
  unsigned char* Ms = (unsigned char*)Ms4;
  unsigned char* Pk = (unsigned char*)Pk4;

  const int tid  = threadIdx.x;
  const int wave = tid >> 5;
  const int lane = tid & 31;
  const int nlo  = lane & 15;
  const int hi   = lane >> 4;
  const int qb = blockIdx.x, h = blockIdx.y, b = blockIdx.z;

  // stage Q tile (128 rows x 64 d) -- pure f16 copy, async
#pragma unroll
  for (int j = 0; j < 4; ++j) {
    int lin = tid + 256 * j, row = lin >> 3, c = lin & 7;
    int s = qb * 128 + row;
    async_copy_b128(&Qs[row * 72 + c * 8],
                    q16 + ((size_t)s * Bq + b) * E + h * 64 + c * 8);
  }
  wait_async0();

  float m[8], l[8];
#pragma unroll
  for (int v = 0; v < 8; ++v) { m[v] = -1e30f; l[v] = 0.f; }
  v8f acc[4];
  const v8f zero8 = {0.f,0.f,0.f,0.f,0.f,0.f,0.f,0.f};
#pragma unroll
  for (int nf = 0; nf < 4; ++nf) acc[nf] = zero8;

  for (int kb = 0; kb < S / 64; ++kb) {
    __syncthreads();   // previous-iteration readers done; Qs visible (kb==0)

    // stage K block transposed: Ks[d][key] (needs transpose -> manual)
#pragma unroll
    for (int j = 0; j < 2; ++j) {
      int lin = tid + 256 * j, key = lin >> 3, c = lin & 7;
      int sk = kb * 64 + key;
      h8 kv = *(const h8*)(k16 + ((size_t)sk * Bq + b) * E + h * 64 + c * 8);
#pragma unroll
      for (int e = 0; e < 8; ++e) Ks[(c * 8 + e) * 72 + key] = kv[e];
    }
    // stage V block (byte-exact) + masks: async
#pragma unroll
    for (int j = 0; j < 2; ++j) {
      int lin = tid + 256 * j, key = lin >> 3, c = lin & 7;
      int sk = kb * 64 + key;
      async_copy_b128(&Vs[key * 72 + c * 8],
                      v16 + ((size_t)sk * Bq + b) * E + h * 64 + c * 8);
    }
#pragma unroll
    for (int j = 0; j < 2; ++j) {
      int lin = tid + 256 * j, row = lin >> 2, c = lin & 3;
      async_copy_b128(&Ms4[lin],
                      amask + (size_t)(qb * 128 + row) * S + kb * 64 + c * 16);
    }
    if (tid < 4)
      async_copy_b128(&Pk4[tid], pmask + (size_t)b * S + kb * 64 + tid * 16);
    wait_async0();
    __syncthreads();

    // scores = Q (16x64) @ K^T (64x64), per wave
    v8f sc[4];
#pragma unroll
    for (int nf = 0; nf < 4; ++nf) sc[nf] = zero8;
#pragma unroll
    for (int ks = 0; ks < 2; ++ks) {
      v16h aq = load_a_frag(Qs + wave * 16 * 72 + ks * 32, 72);
#pragma unroll
      for (int nf = 0; nf < 4; ++nf) {
        v16h bk = load_b_frag(Ks + ks * 32 * 72 + nf * 16, 72);
        sc[nf] = wmma16(aq, bk, sc[nf]);
      }
    }

    // mask + per-row tile max
    float mloc[8];
#pragma unroll
    for (int v = 0; v < 8; ++v) mloc[v] = -1e30f;
#pragma unroll
    for (int nf = 0; nf < 4; ++nf) {
      const int key = nf * 16 + nlo;
      const bool pm = Pk[key] != 0;
#pragma unroll
      for (int v = 0; v < 8; ++v) {
        const int row = wave * 16 + v + 8 * hi;
        const bool mk = pm || (Ms[row * 64 + key] != 0);
        float s = mk ? -1e30f : sc[nf][v];
        sc[nf][v] = s;
        mloc[v] = fmaxf(mloc[v], s);
      }
    }
    // reduce across 16 lanes (xor<16 stays inside each half; rows depend on hi)
#pragma unroll
    for (int xm = 1; xm < 16; xm <<= 1)
#pragma unroll
      for (int v = 0; v < 8; ++v)
        mloc[v] = fmaxf(mloc[v], __shfl_xor(mloc[v], xm, 32));

    float rs[8], ssum[8];
#pragma unroll
    for (int v = 0; v < 8; ++v) {
      float mn = fmaxf(m[v], mloc[v]);
      rs[v] = __expf(m[v] - mn);
      m[v] = mn;
      ssum[v] = 0.f;
    }
#pragma unroll
    for (int nf = 0; nf < 4; ++nf)
#pragma unroll
      for (int v = 0; v < 8; ++v) {
        float p = __expf(sc[nf][v] - m[v]);
        sc[nf][v] = p;
        ssum[v] += p;
      }
#pragma unroll
    for (int xm = 1; xm < 16; xm <<= 1)
#pragma unroll
      for (int v = 0; v < 8; ++v)
        ssum[v] += __shfl_xor(ssum[v], xm, 32);
#pragma unroll
    for (int v = 0; v < 8; ++v) l[v] = l[v] * rs[v] + ssum[v];
#pragma unroll
    for (int nf = 0; nf < 4; ++nf)
#pragma unroll
      for (int v = 0; v < 8; ++v) acc[nf][v] *= rs[v];

    // re-layout P (C-frags) -> LDS -> A-frags.  Ps is PER-WAVE: same-wave LDS
    // RAW only -> scheduling fence is enough (DS is in-order per wave, and the
    // backend inserts the s_wait_dscnt for the aliasing reads).
    _Float16* PsW = Ps + wave * 16 * 72;
#pragma unroll
    for (int nf = 0; nf < 4; ++nf)
#pragma unroll
      for (int v = 0; v < 8; ++v)
        PsW[(v + 8 * hi) * 72 + nf * 16 + nlo] = (_Float16)sc[nf][v];
    __builtin_amdgcn_wave_barrier();

    // out += P (16x64) @ V (64x64)
#pragma unroll
    for (int ks = 0; ks < 2; ++ks) {
      v16h ap = load_a_frag(PsW + ks * 32, 72);
#pragma unroll
      for (int nf = 0; nf < 4; ++nf) {
        v16h bv = load_b_frag(Vs + ks * 32 * 72 + nf * 16, 72);
        acc[nf] = wmma16(ap, bv, acc[nf]);
      }
    }
  }

  // finalize: divide by l, store f16 in (S,B,E) layout
#pragma unroll
  for (int nf = 0; nf < 4; ++nf)
#pragma unroll
    for (int v = 0; v < 8; ++v) {
      const int row = qb * 128 + wave * 16 + v + 8 * hi;
      float o = acc[nf][v] / l[v];
      o16[((size_t)row * Bq + b) * E + h * 64 + nf * 16 + nlo] = (_Float16)o;
    }
}

// ---------------------------------------------------------------------------
// LayerNorm over E=768: one block (256 thr) per row; optional f16 copy.
// ---------------------------------------------------------------------------
template<bool W16>
__global__ __launch_bounds__(256) void ln_kernel(
    const float* __restrict__ y, const float* __restrict__ g,
    const float* __restrict__ bb, float* __restrict__ out,
    _Float16* __restrict__ out16)
{
  constexpr int E = 768;
  const size_t row = blockIdx.x;
  const float* yr = y + row * E;
  const int t = threadIdx.x;
  float v0 = yr[t], v1 = yr[t + 256], v2 = yr[t + 512];
  float s  = v0 + v1 + v2;
  float s2 = v0 * v0 + v1 * v1 + v2 * v2;
#pragma unroll
  for (int xm = 1; xm < 32; xm <<= 1) {
    s  += __shfl_xor(s,  xm, 32);
    s2 += __shfl_xor(s2, xm, 32);
  }
  __shared__ float as_[8], aq_[8];
  const int wave = t >> 5, lane = t & 31;
  if (lane == 0) { as_[wave] = s; aq_[wave] = s2; }
  __syncthreads();
  float ts = 0.f, tq = 0.f;
#pragma unroll
  for (int i = 0; i < 8; ++i) { ts += as_[i]; tq += aq_[i]; }
  const float mu   = ts * (1.f / 768.f);
  const float var  = tq * (1.f / 768.f) - mu * mu;
  const float rstd = rsqrtf(var + 1e-5f);
  const float vv[3] = { v0, v1, v2 };
#pragma unroll
  for (int k = 0; k < 3; ++k) {
    const int c = t + k * 256;
    float o = (vv[k] - mu) * rstd * g[c] + bb[c];
    out[row * E + c] = o;
    if (W16) out16[row * E + c] = (_Float16)o;
  }
}

// ---------------------------------------------------------------------------
// Host launcher
// ---------------------------------------------------------------------------
extern "C" void kernel_launch(void* const* d_in, const int* in_sizes, int n_in,
                              void* d_out, int out_size, void* d_ws, size_t ws_size,
                              hipStream_t stream) {
  (void)in_sizes; (void)n_in; (void)out_size; (void)ws_size;
  constexpr int S = 2048, Bq = 4, E = 768, F = 3072, H = 12;
  const int M = S * Bq;                       // 8192

  const float* x      = (const float*)d_in[0];
  const unsigned char* amask = (const unsigned char*)d_in[1];
  const unsigned char* pmask = (const unsigned char*)d_in[2];
  const float* Wq = (const float*)d_in[3];  const float* bq = (const float*)d_in[4];
  const float* Wk = (const float*)d_in[5];  const float* bk = (const float*)d_in[6];
  const float* Wv = (const float*)d_in[7];  const float* bv = (const float*)d_in[8];
  const float* Wo = (const float*)d_in[9];  const float* bo = (const float*)d_in[10];
  const float* ln1g = (const float*)d_in[11]; const float* ln1b = (const float*)d_in[12];
  const float* W1 = (const float*)d_in[13]; const float* b1 = (const float*)d_in[14];
  const float* W2 = (const float*)d_in[15]; const float* b2 = (const float*)d_in[16];
  const float* ln2g = (const float*)d_in[17]; const float* ln2b = (const float*)d_in[18];

  // Workspace layout (with reuse):
  //   region0: q16|k16|v16|attn16 (4 * M*E*2) -> later reused as hbuf (M*F*2)
  //   region1: y1 (M*E*4)                     -> later reused as y2
  //   region2: x1 (M*E*4);  region3: x1h (M*E*2)
  char* ws = (char*)d_ws;
  const size_t QB = (size_t)M * E * sizeof(_Float16);
  const size_t RB = (size_t)M * E * sizeof(float);
  _Float16* q16    = (_Float16*)(ws);
  _Float16* k16    = (_Float16*)(ws + QB);
  _Float16* v16    = (_Float16*)(ws + 2 * QB);
  _Float16* attn16 = (_Float16*)(ws + 3 * QB);
  _Float16* hbuf   = (_Float16*)(ws);                   // overlays q/k/v/attn
  float*    y1     = (float*)   (ws + 4 * QB);
  float*    y2     = y1;                                // overlays y1
  float*    x1     = (float*)   (ws + 4 * QB + RB);
  _Float16* x1h    = (_Float16*)(ws + 4 * QB + 2 * RB);

  const dim3 blk(256);
  const dim3 gE(E / 128, M / 128);   // (6, 64)
  const dim3 gF(F / 128, M / 128);   // (24, 64)

  // 1) QKV projections (q pre-scaled by Dh^-0.5 = 0.125)
  gemm_wmma<false, EPI_F16_SCALE><<<gE, blk, 0, stream>>>(
      x, Wq, bq, nullptr, 0.125f, q16, nullptr, M, E, E);
  gemm_wmma<false, EPI_F16_SCALE><<<gE, blk, 0, stream>>>(
      x, Wk, bk, nullptr, 1.0f, k16, nullptr, M, E, E);
  gemm_wmma<false, EPI_F16_SCALE><<<gE, blk, 0, stream>>>(
      x, Wv, bv, nullptr, 1.0f, v16, nullptr, M, E, E);

  // 2) attention (flash, per (b,h,128-q-rows))
  attn_kernel<<<dim3(S / 128, H, Bq), blk, 0, stream>>>(
      q16, k16, v16, amask, pmask, attn16);

  // 3) out projection + residual -> y1 (f32)
  gemm_wmma<true, EPI_F32_RES><<<gE, blk, 0, stream>>>(
      attn16, Wo, bo, x, 1.0f, nullptr, y1, M, E, E);

  // 4) LayerNorm1 -> x1 (f32) + x1h (f16)
  ln_kernel<true><<<dim3(M), blk, 0, stream>>>(y1, ln1g, ln1b, x1, x1h);

  // 5) FFN1 + exact GELU -> hbuf (f16)
  gemm_wmma<true, EPI_GELU_F16><<<gF, blk, 0, stream>>>(
      x1h, W1, b1, nullptr, 1.0f, hbuf, nullptr, M, F, E);

  // 6) FFN2 + residual -> y2 (f32)
  gemm_wmma<true, EPI_F32_RES><<<gE, blk, 0, stream>>>(
      hbuf, W2, b2, x1, 1.0f, nullptr, y2, M, E, F);

  // 7) LayerNorm2 -> final output (f32)
  ln_kernel<false><<<dim3(M), blk, 0, stream>>>(
      y2, ln2g, ln2b, (float*)d_out, nullptr);
}
